// EMPSN_30863634989079
// MI455X (gfx1250) — compile-verified, matrix operand order checked by reading, not run on hardware
//
#include <hip/hip_runtime.h>

typedef __attribute__((ext_vector_type(16))) __bf16 v16bf;
typedef __attribute__((ext_vector_type(8)))  float  v8f;

// ---------------- common device helpers ----------------

__device__ __forceinline__ int kmap_i(int i, int khalf) {
  // element i (0..15) of a 16-bit WMMA fragment, half-wave khalf (0/1)
  // -> K position inside a 32-deep chunk  (ISA 7.12.2, 16-bit A 16x32)
  return ((i >> 3) << 4) + (khalf << 3) + (i & 7);
}

__device__ __forceinline__ float siluf(float v) { return v * (1.0f / (1.0f + __expf(-v))); }
__device__ __forceinline__ float sigmf(float v) { return 1.0f / (1.0f + __expf(-v)); }

__device__ __forceinline__ void atomAddGlobal(float* p, float v) {
  __hip_atomic_fetch_add(p, v, __ATOMIC_RELAXED, __HIP_MEMORY_SCOPE_AGENT);
}
__device__ __forceinline__ void atomAddShared(float* p, float v) {
  __hip_atomic_fetch_add(p, v, __ATOMIC_RELAXED, __HIP_MEMORY_SCOPE_WORKGROUP);
}

__device__ __forceinline__ v8f wmma_step(const __bf16* Afrag, const __bf16* Bfrag, v8f acc) {
  v16bf a = *(const v16bf*)Afrag;
  v16bf b = *(const v16bf*)Bfrag;
  return __builtin_amdgcn_wmma_f32_16x16x32_bf16(false, a, false, b, (short)0, acc, false, false);
}

// store an fp32 value at logical (row, col) into an LDS fragment buffer (bf16)
__device__ __forceinline__ void storeFrag(__bf16* buf, int row, int col, float v) {
  int ktile = col >> 5, kin = col & 31;
  int i    = ((kin >> 4) << 3) | (kin & 7);
  int lane = (((kin >> 3) & 1) << 4) + row;
  buf[(ktile * 32 + lane) * 16 + i] = (__bf16)v;
}

struct F3 { float x, y, z; };
__device__ __forceinline__ F3 ldp(const float* pos, int i) {
  F3 p; p.x = pos[3 * i]; p.y = pos[3 * i + 1]; p.z = pos[3 * i + 2]; return p;
}
__device__ __forceinline__ F3 sub3(F3 a, F3 b) { F3 r{a.x - b.x, a.y - b.y, a.z - b.z}; return r; }
__device__ __forceinline__ float dot3(F3 a, F3 b) { return a.x * b.x + a.y * b.y + a.z * b.z; }
__device__ __forceinline__ float nrm3(F3 a) { return sqrtf(dot3(a, a) + 1e-12f); }
__device__ __forceinline__ float distp(const float* pos, int a, int b) {
  return nrm3(sub3(ldp(pos, a), ldp(pos, b)));
}
__device__ __forceinline__ float angf(F3 u, F3 v, float un, float vn) {
  float c = dot3(u, v) / (un * vn);
  c = fminf(fmaxf(c, -1.0f + 1e-6f), 1.0f - 1e-6f);
  return acosf(c);
}

// ---------------- weight swizzle: fp32 [K][128] -> bf16 fragment layout ----------------

__global__ void __launch_bounds__(256)
swizzle_w_kernel(const float* __restrict__ W, int K, int Kpad, __bf16* __restrict__ out) {
  int o = blockIdx.x * 256 + threadIdx.x;
  if (o >= Kpad * 128) return;
  int i = o & 15;
  int lane = (o >> 4) & 31;
  int ntile = (o >> 9) & 7;
  int ktile = o >> 12;
  int k = ktile * 32 + kmap_i(i, lane >> 4);
  int col = ntile * 16 + (lane & 15);
  float v = (k < K) ? W[(size_t)k * 128 + col] : 0.0f;
  out[o] = (__bf16)v;
}

// ---------------- embed: mean of vertex feats, then 15->128 linear ----------------

__global__ void __launch_bounds__(256)
embed_kernel(const float* __restrict__ x, const int* __restrict__ ind, int nvert, int N,
             const float* __restrict__ W, const float* __restrict__ b, float* __restrict__ out) {
  __shared__ float xm[2][15];
  int tid = threadIdx.x;
  int nb = blockIdx.x * 2;
  if (tid < 30) {
    int ls = tid / 15, k = tid % 15;
    int n = min(nb + ls, N - 1);
    float s = 0.0f;
    for (int v = 0; v < nvert; ++v) s += x[(size_t)ind[n * nvert + v] * 15 + k];
    xm[ls][k] = s / (float)nvert;
  }
  __syncthreads();
  int ls = tid >> 7, c = tid & 127;
  int n = nb + ls;
  if (n < N) {
    float acc = b[c];
    for (int k = 0; k < 15; ++k) acc += xm[ls][k] * W[k * 128 + c];
    out[(size_t)n * 128 + c] = acc;
  }
}

// ---------------- invariants ----------------

__global__ void area_edge_kernel(const float* __restrict__ pos, const int* __restrict__ x1,
                                 int N, float* __restrict__ area) {
  int n = blockIdx.x * 256 + threadIdx.x;
  if (n >= N) return;
  area[n] = distp(pos, x1[n * 2], x1[n * 2 + 1]);
}

__global__ void area_tri_kernel(const float* __restrict__ pos, const int* __restrict__ x2,
                                int N, float* __restrict__ area) {
  int n = blockIdx.x * 256 + threadIdx.x;
  if (n >= N) return;
  F3 a = ldp(pos, x2[n * 3]), b = ldp(pos, x2[n * 3 + 1]), c = ldp(pos, x2[n * 3 + 2]);
  F3 u = sub3(a, b), v = sub3(a, c);
  F3 cr{u.y * v.z - u.z * v.y, u.z * v.x - u.x * v.z, u.x * v.y - u.y * v.x};
  area[n] = 0.5f * nrm3(cr);
}

__global__ void inv00_kernel(const float* __restrict__ pos, const int* __restrict__ s,
                             const int* __restrict__ r, int E, float* __restrict__ out) {
  int e = blockIdx.x * 256 + threadIdx.x;
  if (e >= E) return;
  out[e * 3 + 0] = distp(pos, s[e], r[e]);
  out[e * 3 + 1] = 0.0f;
  out[e * 3 + 2] = 0.0f;
}

__global__ void inv01_kernel(const float* __restrict__ pos, const int* __restrict__ s,
                             const int* __restrict__ r, const int* __restrict__ i0,
                             const int* __restrict__ i1, const float* __restrict__ area1,
                             int E, float* __restrict__ out) {
  int e = blockIdx.x * 256 + threadIdx.x;
  if (e >= E) return;
  out[e * 3 + 0] = distp(pos, s[e], i0[e]) + distp(pos, s[e], i1[e]);
  out[e * 3 + 1] = 0.0f;
  out[e * 3 + 2] = area1[r[e]];
}

__global__ void inv11_kernel(const float* __restrict__ pos, const int* __restrict__ x1,
                             const int* __restrict__ s, const int* __restrict__ r,
                             const int* __restrict__ i0, const int* __restrict__ i1,
                             const int* __restrict__ i2, const float* __restrict__ area1,
                             int E, float* __restrict__ out) {
  int e = blockIdx.x * 256 + threadIdx.x;
  if (e >= E) return;
  F3 sv = sub3(ldp(pos, x1[s[e] * 2]), ldp(pos, x1[s[e] * 2 + 1]));
  F3 rv = sub3(ldp(pos, x1[r[e] * 2]), ldp(pos, x1[r[e] * 2 + 1]));
  float ang = angf(sv, rv, nrm3(sv), nrm3(rv));
  out[e * 6 + 0] = distp(pos, i0[e], i1[e]);
  out[e * 6 + 1] = distp(pos, i0[e], i2[e]);
  out[e * 6 + 2] = distp(pos, i1[e], i2[e]);
  out[e * 6 + 3] = area1[s[e]];
  out[e * 6 + 4] = area1[r[e]];
  out[e * 6 + 5] = ang;
}

__global__ void inv12_kernel(const float* __restrict__ pos, const int* __restrict__ s,
                             const int* __restrict__ r, const int* __restrict__ i0,
                             const int* __restrict__ i1, const int* __restrict__ i2,
                             const float* __restrict__ area1, const float* __restrict__ area2,
                             int E, float* __restrict__ out) {
  int e = blockIdx.x * 256 + threadIdx.x;
  if (e >= E) return;
  F3 p1 = ldp(pos, i0[e]), p2 = ldp(pos, i1[e]), a = ldp(pos, i2[e]);
  F3 v1 = sub3(p1, a), v2 = sub3(p2, a), bb = sub3(p1, p2);
  float v1n = nrm3(v1), v2n = nrm3(v2), bn = nrm3(bb);
  out[e * 6 + 0] = v1n + v2n;   // d(i0,i2)+d(i1,i2)
  out[e * 6 + 1] = bn;          // d(i0,i1)
  out[e * 6 + 2] = area1[s[e]];
  out[e * 6 + 3] = area2[r[e]];
  out[e * 6 + 4] = angf(v1, bb, v1n, bn) + angf(v2, bb, v2n, bn);
  out[e * 6 + 5] = angf(v1, v2, v1n, v2n);
}

// ---------------- fused message kernel (gather -> MLP(WMMA) -> gate -> scatter) ----------------
// block = 256 threads = 8 waves; tile = 16 edges; wave w handles output cols [16w,16w+16)

__global__ void __launch_bounds__(256)
msg_kernel(const float* __restrict__ fs, const float* __restrict__ fr,
           const int* __restrict__ idxS, const int* __restrict__ idxR, int E,
           const float* __restrict__ inv, int ninv,
           const __bf16* __restrict__ W1s, const float* __restrict__ b1,
           const __bf16* __restrict__ W2s, const float* __restrict__ b2,
           const float* __restrict__ We, const float* __restrict__ be,
           float* __restrict__ mes) {
  __shared__ __align__(32) __bf16 Abuf[9 * 512];   // K=288 in 9 fragment chunks
  __shared__ __align__(32) __bf16 A2buf[4 * 512];  // K=128 hidden
  __shared__ int sIs[16], sIr[16];
  __shared__ float rowacc[16];
  const int tid = threadIdx.x, lane = tid & 31, wv = tid >> 5;
  const int e0 = blockIdx.x * 16;
  if (tid < 16) {
    int e = min(e0 + tid, E - 1);
    sIs[tid] = idxS[e];
    sIr[tid] = idxR[e];
    rowacc[tid] = 0.0f;
  }
  __syncthreads();
  // gather state = [feats_s | feats_r | inv | 0-pad] into fragment layout
  for (int f = tid; f < 9 * 512; f += 256) {
    int i = f & 15, ln = (f >> 4) & 31, kt = f >> 9;
    int k = kt * 32 + kmap_i(i, ln >> 4);
    int row = ln & 15;
    float v = 0.0f;
    if (k < 128)            v = fs[(size_t)sIs[row] * 128 + k];
    else if (k < 256)       v = fr[(size_t)sIr[row] * 128 + (k - 128)];
    else if (k < 256 + ninv) {
      int e = min(e0 + row, E - 1);
      v = inv[(size_t)e * ninv + (k - 256)];
    }
    Abuf[f] = (__bf16)v;
  }
  __syncthreads();
  v8f acc = {};
  #pragma unroll
  for (int kk = 0; kk < 9; ++kk)
    acc = wmma_step(&Abuf[(kk * 32 + lane) * 16], &W1s[((kk * 8 + wv) * 32 + lane) * 16], acc);
  const int col = wv * 16 + (lane & 15);
  float bb1 = b1[col];
  #pragma unroll
  for (int r = 0; r < 8; ++r) {
    int row = ((lane >> 4) << 3) + r;
    storeFrag(A2buf, row, col, siluf(acc[r] + bb1));
  }
  __syncthreads();
  v8f acc2 = {};
  #pragma unroll
  for (int kk = 0; kk < 4; ++kk)
    acc2 = wmma_step(&A2buf[(kk * 32 + lane) * 16], &W2s[((kk * 8 + wv) * 32 + lane) * 16], acc2);
  float bb2 = b2[col], we = We[col];
  float mv[8];
  #pragma unroll
  for (int r = 0; r < 8; ++r) {
    int row = ((lane >> 4) << 3) + r;
    float v = siluf(acc2[r] + bb2);
    mv[r] = v;
    atomAddShared(&rowacc[row], v * we);   // einf dot-product partial
  }
  __syncthreads();
  float beV = be[0];
  #pragma unroll
  for (int r = 0; r < 8; ++r) {
    int row = ((lane >> 4) << 3) + r;
    if (e0 + row < E) {
      float g = sigmf(rowacc[row] + beV);
      atomAddGlobal(&mes[(size_t)sIr[row] * 128 + col], mv[r] * g);
    }
  }
}

// ---------------- update kernel: residual MLP over [feats | mes...] ----------------

__global__ void __launch_bounds__(256)
upd_kernel(const float* __restrict__ feats, const float* __restrict__ mesA,
           const float* __restrict__ mesB, int N, int nseg,
           const __bf16* __restrict__ W1s, const float* __restrict__ b1,
           const __bf16* __restrict__ W2s, const float* __restrict__ b2,
           float* __restrict__ out) {
  __shared__ __align__(32) __bf16 Abuf[12 * 512];  // up to K=384
  __shared__ __align__(32) __bf16 A2buf[4 * 512];
  const int tid = threadIdx.x, lane = tid & 31, wv = tid >> 5;
  const int n0 = blockIdx.x * 16;
  const int chunks = nseg * 4;
  for (int f = tid; f < chunks * 512; f += 256) {
    int i = f & 15, ln = (f >> 4) & 31, kt = f >> 9;
    int k = kt * 32 + kmap_i(i, ln >> 4);
    int row = ln & 15;
    int n = min(n0 + row, N - 1);
    float v;
    if (k < 128)      v = feats[(size_t)n * 128 + k];
    else if (k < 256) v = mesA[(size_t)n * 128 + (k - 128)];
    else              v = mesB[(size_t)n * 128 + (k - 256)];
    Abuf[f] = (__bf16)v;
  }
  __syncthreads();
  v8f acc = {};
  for (int kk = 0; kk < chunks; ++kk)
    acc = wmma_step(&Abuf[(kk * 32 + lane) * 16], &W1s[((kk * 8 + wv) * 32 + lane) * 16], acc);
  const int col = wv * 16 + (lane & 15);
  float bb1 = b1[col];
  #pragma unroll
  for (int r = 0; r < 8; ++r) {
    int row = ((lane >> 4) << 3) + r;
    storeFrag(A2buf, row, col, siluf(acc[r] + bb1));
  }
  __syncthreads();
  v8f acc2 = {};
  #pragma unroll
  for (int kk = 0; kk < 4; ++kk)
    acc2 = wmma_step(&A2buf[(kk * 32 + lane) * 16], &W2s[((kk * 8 + wv) * 32 + lane) * 16], acc2);
  float bb2 = b2[col];
  #pragma unroll
  for (int r = 0; r < 8; ++r) {
    int row = ((lane >> 4) << 3) + r;
    int n = n0 + row;
    if (n < N) out[(size_t)n * 128 + col] = feats[(size_t)n * 128 + col] + acc2[r] + bb2;
  }
}

// ---------------- pool kernel: pre-MLP + batch segment-sum ----------------

__global__ void __launch_bounds__(256)
pool_kernel(const float* __restrict__ feats, int N, const int* __restrict__ batch,
            const __bf16* __restrict__ W1s, const float* __restrict__ b1,
            const __bf16* __restrict__ W2s, const float* __restrict__ b2,
            float* __restrict__ pooled, int coff) {
  __shared__ __align__(32) __bf16 Abuf[4 * 512];
  __shared__ __align__(32) __bf16 A2buf[4 * 512];
  __shared__ int sB[16];
  const int tid = threadIdx.x, lane = tid & 31, wv = tid >> 5;
  const int n0 = blockIdx.x * 16;
  if (tid < 16) sB[tid] = batch[min(n0 + tid, N - 1)];
  __syncthreads();
  for (int f = tid; f < 4 * 512; f += 256) {
    int i = f & 15, ln = (f >> 4) & 31, kt = f >> 9;
    int k = kt * 32 + kmap_i(i, ln >> 4);
    int row = ln & 15;
    int n = min(n0 + row, N - 1);
    Abuf[f] = (__bf16)feats[(size_t)n * 128 + k];
  }
  __syncthreads();
  v8f acc = {};
  #pragma unroll
  for (int kk = 0; kk < 4; ++kk)
    acc = wmma_step(&Abuf[(kk * 32 + lane) * 16], &W1s[((kk * 8 + wv) * 32 + lane) * 16], acc);
  const int col = wv * 16 + (lane & 15);
  float bb1 = b1[col];
  #pragma unroll
  for (int r = 0; r < 8; ++r) {
    int row = ((lane >> 4) << 3) + r;
    storeFrag(A2buf, row, col, siluf(acc[r] + bb1));
  }
  __syncthreads();
  v8f acc2 = {};
  #pragma unroll
  for (int kk = 0; kk < 4; ++kk)
    acc2 = wmma_step(&A2buf[(kk * 32 + lane) * 16], &W2s[((kk * 8 + wv) * 32 + lane) * 16], acc2);
  float bb2 = b2[col];
  #pragma unroll
  for (int r = 0; r < 8; ++r) {
    int row = ((lane >> 4) << 3) + r;
    int n = n0 + row;
    if (n < N) atomAddGlobal(&pooled[(size_t)sB[row] * 384 + coff + col], acc2[r] + bb2);
  }
}

// ---------------- post MLP: [B,384] -> [B] ----------------

__global__ void __launch_bounds__(128)
post_kernel(const float* __restrict__ pooled, const float* __restrict__ q1W,
            const float* __restrict__ q1b, const float* __restrict__ q2W,
            const float* __restrict__ q2b, float* __restrict__ out) {
  __shared__ float red[128];
  int b = blockIdx.x, c = threadIdx.x;
  const float* st = pooled + (size_t)b * 384;
  float acc = q1b[c];
  for (int k = 0; k < 384; ++k) acc += st[k] * q1W[k * 128 + c];
  red[c] = siluf(acc) * q2W[c];
  __syncthreads();
  for (int s = 64; s > 0; s >>= 1) {
    if (c < s) red[c] += red[c + s];
    __syncthreads();
  }
  if (c == 0) out[b] = red[0] + q2b[0];
}

// ---------------- host orchestration ----------------

extern "C" void kernel_launch(void* const* d_in, const int* in_sizes, int n_in,
                              void* d_out, int out_size, void* d_ws, size_t ws_size,
                              hipStream_t stream) {
  (void)in_sizes; (void)n_in; (void)out_size; (void)ws_size;
  constexpr int N0 = 20000, N1 = 60000, N2 = 20000;
  constexpr int E00 = 320000, E01 = 120000, E11 = 400000, E12 = 60000;
  constexpr int NB = 64;

  const float* x   = (const float*)d_in[0];
  const float* pos = (const float*)d_in[1];
  const int* x0a = (const int*)d_in[2];
  const int* x1a = (const int*)d_in[3];
  const int* x2a = (const int*)d_in[4];
  const int* b0a = (const int*)d_in[5];
  const int* b1a = (const int*)d_in[6];
  const int* b2a = (const int*)d_in[7];
  const int* a00 = (const int*)d_in[8];
  const int* a01 = (const int*)d_in[9];
  const int* a11 = (const int*)d_in[10];
  const int* a12 = (const int*)d_in[11];
  const int* i01 = (const int*)d_in[12];
  const int* i11 = (const int*)d_in[13];
  const int* i12 = (const int*)d_in[14];
  // params flattened (jax pytree sorted-key order):
  // 0:embed.W 1:embed.b ; layer L adj j: 2+36L+6j -> einf.W,einf.b,m1.W,m1.b,m2.W,m2.b
  // layer L dim d: 2+36L+24+4d -> u1.W,u1.b,u2.W,u2.b ; post: 74..77 ; pre d: 78+4d..
  auto P = [&](int i) -> const float* { return (const float*)d_in[15 + i]; };

  char* ws = (char*)d_ws;
  size_t off = 0;
  auto take = [&](size_t bytes) -> void* {
    void* p = ws + off;
    off = (off + bytes + 255) & ~(size_t)255;
    return p;
  };
  float* fA[3] = {(float*)take((size_t)N0 * 128 * 4), (float*)take((size_t)N1 * 128 * 4),
                  (float*)take((size_t)N2 * 128 * 4)};
  float* fB[3] = {(float*)take((size_t)N0 * 128 * 4), (float*)take((size_t)N1 * 128 * 4),
                  (float*)take((size_t)N2 * 128 * 4)};
  float* mes00 = (float*)take((size_t)N0 * 128 * 4);
  float* mes01 = (float*)take((size_t)N1 * 128 * 4);
  float* mes11 = (float*)take((size_t)N1 * 128 * 4);
  float* mes12 = (float*)take((size_t)N2 * 128 * 4);
  float* inv00b = (float*)take((size_t)E00 * 3 * 4);
  float* inv01b = (float*)take((size_t)E01 * 3 * 4);
  float* inv11b = (float*)take((size_t)E11 * 6 * 4);
  float* inv12b = (float*)take((size_t)E12 * 6 * 4);
  float* area1b = (float*)take((size_t)N1 * 4);
  float* area2b = (float*)take((size_t)N2 * 4);
  float* pooled = (float*)take((size_t)NB * 384 * 4);
  __bf16 *m1s[2][4], *m2s[2][4], *u1s[2][3], *u2s[2][3], *p1s[3], *p2s[3];
  for (int L = 0; L < 2; ++L)
    for (int j = 0; j < 4; ++j) {
      m1s[L][j] = (__bf16*)take((size_t)288 * 128 * 2);
      m2s[L][j] = (__bf16*)take((size_t)128 * 128 * 2);
    }
  for (int L = 0; L < 2; ++L)
    for (int d = 0; d < 3; ++d) {
      int ns = (d == 1) ? 3 : 2;
      u1s[L][d] = (__bf16*)take((size_t)ns * 128 * 128 * 2);
      u2s[L][d] = (__bf16*)take((size_t)128 * 128 * 2);
    }
  for (int d = 0; d < 3; ++d) {
    p1s[d] = (__bf16*)take((size_t)128 * 128 * 2);
    p2s[d] = (__bf16*)take((size_t)128 * 128 * 2);
  }

  auto swz = [&](const float* W, int K, int Kpad, __bf16* out) {
    int total = Kpad * 128;
    swizzle_w_kernel<<<(total + 255) / 256, 256, 0, stream>>>(W, K, Kpad, out);
  };
  for (int L = 0; L < 2; ++L) {
    for (int j = 0; j < 4; ++j) {
      int ninv = (j < 2) ? 3 : 6;
      int base = 2 + L * 36 + j * 6;
      swz(P(base + 2), 256 + ninv, 288, m1s[L][j]);
      swz(P(base + 4), 128, 128, m2s[L][j]);
    }
    for (int d = 0; d < 3; ++d) {
      int base = 2 + L * 36 + 24 + d * 4;
      int ns = (d == 1) ? 3 : 2;
      swz(P(base + 0), ns * 128, ns * 128, u1s[L][d]);
      swz(P(base + 2), 128, 128, u2s[L][d]);
    }
  }
  for (int d = 0; d < 3; ++d) {
    int pb = 78 + d * 4;
    swz(P(pb + 0), 128, 128, p1s[d]);
    swz(P(pb + 2), 128, 128, p2s[d]);
  }

  // embed
  embed_kernel<<<(N0 + 1) / 2, 256, 0, stream>>>(x, x0a, 1, N0, P(0), P(1), fA[0]);
  embed_kernel<<<(N1 + 1) / 2, 256, 0, stream>>>(x, x1a, 2, N1, P(0), P(1), fA[1]);
  embed_kernel<<<(N2 + 1) / 2, 256, 0, stream>>>(x, x2a, 3, N2, P(0), P(1), fA[2]);

  // invariants
  area_edge_kernel<<<(N1 + 255) / 256, 256, 0, stream>>>(pos, x1a, N1, area1b);
  area_tri_kernel<<<(N2 + 255) / 256, 256, 0, stream>>>(pos, x2a, N2, area2b);
  inv00_kernel<<<(E00 + 255) / 256, 256, 0, stream>>>(pos, a00, a00 + E00, E00, inv00b);
  inv01_kernel<<<(E01 + 255) / 256, 256, 0, stream>>>(pos, a01, a01 + E01, i01, i01 + E01,
                                                      area1b, E01, inv01b);
  inv11_kernel<<<(E11 + 255) / 256, 256, 0, stream>>>(pos, x1a, a11, a11 + E11, i11, i11 + E11,
                                                      i11 + 2 * (size_t)E11, area1b, E11, inv11b);
  inv12_kernel<<<(E12 + 255) / 256, 256, 0, stream>>>(pos, a12, a12 + E12, i12, i12 + E12,
                                                      i12 + 2 * (size_t)E12, area1b, area2b, E12,
                                                      inv12b);

  float* cur[3] = {fA[0], fA[1], fA[2]};
  float* nxt[3] = {fB[0], fB[1], fB[2]};
  for (int L = 0; L < 2; ++L) {
    hipMemsetAsync(mes00, 0, (size_t)N0 * 128 * 4, stream);
    hipMemsetAsync(mes01, 0, (size_t)N1 * 128 * 4, stream);
    hipMemsetAsync(mes11, 0, (size_t)N1 * 128 * 4, stream);
    hipMemsetAsync(mes12, 0, (size_t)N2 * 128 * 4, stream);
    struct AdjDesc {
      const float *fs, *fr;
      const int *is, *ir;
      int E;
      const float* inv;
      int ninv;
      float* mes;
    } A[4] = {{cur[0], cur[0], a00, a00 + E00, E00, inv00b, 3, mes00},
              {cur[0], cur[1], a01, a01 + E01, E01, inv01b, 3, mes01},
              {cur[1], cur[1], a11, a11 + E11, E11, inv11b, 6, mes11},
              {cur[1], cur[2], a12, a12 + E12, E12, inv12b, 6, mes12}};
    for (int j = 0; j < 4; ++j) {
      int base = 2 + L * 36 + j * 6;
      msg_kernel<<<(A[j].E + 15) / 16, 256, 0, stream>>>(
          A[j].fs, A[j].fr, A[j].is, A[j].ir, A[j].E, A[j].inv, A[j].ninv,
          m1s[L][j], P(base + 3), m2s[L][j], P(base + 5), P(base + 0), P(base + 1), A[j].mes);
    }
    const float* mA[3] = {mes00, mes01, mes12};
    const float* mB[3] = {mes00, mes11, mes12};  // mB unused for d=0,2
    int Ns[3] = {N0, N1, N2};
    for (int d = 0; d < 3; ++d) {
      int ub = 2 + L * 36 + 24 + d * 4;
      int ns = (d == 1) ? 3 : 2;
      upd_kernel<<<(Ns[d] + 15) / 16, 256, 0, stream>>>(cur[d], mA[d], mB[d], Ns[d], ns,
                                                        u1s[L][d], P(ub + 1), u2s[L][d], P(ub + 3),
                                                        nxt[d]);
    }
    for (int d = 0; d < 3; ++d) {
      float* t = cur[d];
      cur[d] = nxt[d];
      nxt[d] = t;
    }
  }

  hipMemsetAsync(pooled, 0, (size_t)NB * 384 * 4, stream);
  const int* bat[3] = {b0a, b1a, b2a};
  int Ns[3] = {N0, N1, N2};
  for (int d = 0; d < 3; ++d) {
    int pb = 78 + d * 4;
    pool_kernel<<<(Ns[d] + 15) / 16, 256, 0, stream>>>(cur[d], Ns[d], bat[d], p1s[d], P(pb + 1),
                                                       p2s[d], P(pb + 3), pooled, d * 128);
  }
  post_kernel<<<NB, 128, 0, stream>>>(pooled, P(74), P(75), P(76), P(77), (float*)d_out);
}
